// MultiHeadAttention_3547642987057
// MI455X (gfx1250) — compile-verified, matrix operand order checked by reading, not run on hardware
//
#include <hip/hip_runtime.h>
#include <hip/hip_bf16.h>

// ---------------------------------------------------------------------------
// MI455X (gfx1250) multi-head attention forward.
// All GEMMs via v_wmma_f32_16x16x32_bf16 (wave32 WMMA, f32 accumulate).
// Inputs/weights pre-converted to bf16 so GEMM fragment loads are pure dwords.
// ---------------------------------------------------------------------------

typedef __bf16 bf16_t;
typedef __attribute__((ext_vector_type(16))) __bf16 v16bf;
typedef __attribute__((ext_vector_type(2)))  __bf16 v2bf;
typedef __attribute__((ext_vector_type(2)))  float  v2f;
typedef __attribute__((ext_vector_type(8)))  float  v8f;
typedef __attribute__((ext_vector_type(8)))  unsigned v8u;

#define D_MODEL 512
#define NUM_HEADS 8
#define HEAD_DIM 64
#define BATCH 4
#define SEQ 2048
#define MROWS (BATCH * SEQ)          // 8192
#define BH (BATCH * NUM_HEADS)       // 32

// ---- bf16 helpers (native conversion; backend picks v_cvt_pk_bf16_f32) ----

static __device__ inline bf16_t f2bf(float f) { return (bf16_t)f; }

static __device__ inline unsigned pack2bf(float lo, float hi) {
  v2f f = {lo, hi};
  v2bf b = __builtin_convertvector(f, v2bf);
  return __builtin_bit_cast(unsigned, b);
}

// k-index pattern for 16-bit A/B fragments (ISA 7.12.2, 16-bit A 16x32):
// vgpr i in 0..3 -> K = half*8 + 2i ; vgpr i in 4..7 -> K = 16 + half*8 + 2(i-4)
static __device__ inline int kidx(int i, int half) {
  return (i < 4) ? (half * 8 + 2 * i) : (16 + half * 8 + 2 * (i - 4));
}

// Load a 16x32 A (or 32x16 transposed-B) fragment from a bf16 row.
// Per-lane row pointer already applied; pairs are 4-byte aligned.
static __device__ inline v16bf frag_bf16(const bf16_t* rowp, int k0, int lane) {
  const int half = (lane >> 4) & 1;
  v8u u;
#pragma unroll
  for (int i = 0; i < 8; ++i) {
    const int k = k0 + kidx(i, half);
    u[i] = *(const unsigned*)(rowp + k);
  }
  return __builtin_bit_cast(v16bf, u);
}

static __device__ inline v8f wmma_bf16(v16bf a, v16bf b, v8f c) {
  return __builtin_amdgcn_wmma_f32_16x16x32_bf16(
      /*neg_a=*/false, a, /*neg_b=*/false, b,
      /*c_mod=*/(short)0, c, /*reuse_a=*/false, /*reuse_b=*/false);
}

// ---------------------------------------------------------------------------
// Stage 0: vectorized fp32 -> bf16 conversion (float4 -> 4 x bf16 per thread)
// ---------------------------------------------------------------------------
__global__ __launch_bounds__(256) void cvt_f32_bf16_kernel(
    const float* __restrict__ src, bf16_t* __restrict__ dst, int n4) {
  const int i = blockIdx.x * 256 + threadIdx.x;
  if (i >= n4) return;
  const float4 f = ((const float4*)src)[i];
  uint2 o;
  o.x = pack2bf(f.x, f.y);
  o.y = pack2bf(f.z, f.w);
  ((uint2*)dst)[i] = o;
}

// ---------------------------------------------------------------------------
// Stage 1: fused Q/K/V projections.  X [8192,512] @ W^T [512,512] + bias.
// All operands bf16. z==0 -> q_ws [B,H,S,Dh] ; z==1 -> k_ws ; z==2 -> vT_ws
// [B,H,Dh,S].  grid (512, 8, 3), block 128 (4 waves; wave -> one 16-col tile).
// ---------------------------------------------------------------------------
__global__ __launch_bounds__(128) void qkv_proj_kernel(
    const bf16_t* __restrict__ Qb, const bf16_t* __restrict__ Kb,
    const bf16_t* __restrict__ Vb, const bf16_t* __restrict__ wqb,
    const bf16_t* __restrict__ wkb, const bf16_t* __restrict__ wvb,
    const float* __restrict__ bq, const float* __restrict__ bk,
    const float* __restrict__ bv, bf16_t* __restrict__ q_ws,
    bf16_t* __restrict__ k_ws, bf16_t* __restrict__ vT_ws) {
  const int z = blockIdx.z;
  const bf16_t* X  = (z == 0) ? Qb  : (z == 1) ? Kb  : Vb;
  const bf16_t* W  = (z == 0) ? wqb : (z == 1) ? wkb : wvb;
  const float*  Bs = (z == 0) ? bq  : (z == 1) ? bk  : bv;

  const int lane = threadIdx.x & 31;
  const int wid  = threadIdx.x >> 5;
  const int m0 = blockIdx.x * 16;
  const int n0 = blockIdx.y * 64 + wid * 16;
  const int hl = (lane >> 4) & 1;
  const int nl = lane & 15;

  const bf16_t* xrow = X + (size_t)(m0 + nl) * D_MODEL;   // A: lane = row m
  const bf16_t* wrow = W + (size_t)(n0 + nl) * D_MODEL;   // B^T: lane = col n

  v8f acc = {};
#pragma unroll 4
  for (int k0 = 0; k0 < D_MODEL; k0 += 32) {
    v16bf a = frag_bf16(xrow, k0, lane);
    v16bf b = frag_bf16(wrow, k0, lane);
    acc = wmma_bf16(a, b, acc);
  }

  const float bias = Bs[n0 + nl];
  const int col = n0 + nl;
  const int h  = col >> 6;         // head
  const int dh = col & 63;         // dim within head
#pragma unroll
  for (int v = 0; v < 8; ++v) {
    const int row = m0 + v + 8 * hl;      // 0..8191
    const int bi = row >> 11;             // batch
    const int s  = row & 2047;            // seq
    const float val = acc[v] + bias;
    if (z == 2) {
      vT_ws[(((size_t)bi * NUM_HEADS + h) * HEAD_DIM + dh) * SEQ + s] = f2bf(val);
    } else {
      bf16_t* dst = (z == 0) ? q_ws : k_ws;
      dst[(((size_t)bi * NUM_HEADS + h) * SEQ + s) * HEAD_DIM + dh] = f2bf(val);
    }
  }
}

// ---------------------------------------------------------------------------
// Stage 2: per (b,h, 16-query-row tile): scores -> causal softmax -> probs
// (fp32 to d_out, bf16 to LDS) -> ctx = P @ V (bf16 to ctx_ws).
// LDS: fp32 scores [16][2048] (128 KB) + bf16 probs [16][2048] (64 KB) + red.
// CDNA5 allows 320 KB/WG -- this would not fit a CDNA4 CU.
// grid 4096, block 128 (4 waves).
// ---------------------------------------------------------------------------
__global__ __launch_bounds__(128) void attn_kernel(
    const bf16_t* __restrict__ q_ws, const bf16_t* __restrict__ k_ws,
    const bf16_t* __restrict__ vT_ws, float* __restrict__ attn_out,
    bf16_t* __restrict__ ctx_ws) {
  extern __shared__ char smem_raw[];
  float*  S   = (float*)smem_raw;                               // [16][2048]
  bf16_t* Pb  = (bf16_t*)(smem_raw + 16 * SEQ * sizeof(float)); // [16][2048]
  float*  red = (float*)(smem_raw + 16 * SEQ * (sizeof(float) + sizeof(bf16_t)));

  const int tid  = threadIdx.x;
  const int lane = tid & 31;
  const int wid  = tid >> 5;            // 0..3
  const int hl   = (lane >> 4) & 1;
  const int nl   = lane & 15;

  const int id = blockIdx.x;
  const int qt = id & 127;              // query tile
  const int h  = (id >> 7) & 7;
  const int bi = id >> 10;
  const size_t bh = (size_t)bi * NUM_HEADS + h;

  // --- Q fragments for this 16-row tile (K = 0..63) ---
  const bf16_t* qrow = q_ws + (bh * SEQ + (size_t)qt * 16 + nl) * HEAD_DIM;
  const v16bf qa0 = frag_bf16(qrow, 0, lane);
  const v16bf qa1 = frag_bf16(qrow, 32, lane);

  // --- scores: wave `wid` handles key tiles kt = wid, wid+4, ... <= qt ---
  for (int kt = wid; kt <= qt; kt += 4) {
    const bf16_t* krow = k_ws + (bh * SEQ + (size_t)kt * 16 + nl) * HEAD_DIM;
    if (kt + 4 <= qt)
      __builtin_prefetch(k_ws + (bh * SEQ + (size_t)(kt + 4) * 16 + nl) * HEAD_DIM, 0, 1);
    v16bf kb0 = frag_bf16(krow, 0, lane);
    v16bf kb1 = frag_bf16(krow, 32, lane);
    v8f c = {};
    c = wmma_bf16(qa0, kb0, c);
    c = wmma_bf16(qa1, kb1, c);
#pragma unroll
    for (int v = 0; v < 8; ++v) {
      const int m = v + 8 * hl;
      S[m * SEQ + kt * 16 + nl] = c[v] * 0.125f;   // 1/sqrt(64)
    }
  }
  __syncthreads();

  // --- causal softmax over rows; 8 threads per row ---
  const int r   = tid >> 3;             // 0..15
  const int sub = tid & 7;
  const int valid = qt * 16 + r + 1;    // causal row length
  float* Srow = S + r * SEQ;

  float mx = -3.0e38f;
  for (int c = sub; c < valid; c += 8) mx = fmaxf(mx, Srow[c]);
  red[r * 8 + sub] = mx;
  __syncthreads();
#pragma unroll
  for (int j = 0; j < 8; ++j) mx = fmaxf(mx, red[r * 8 + j]);
  float sum = 0.f;
  for (int c = sub; c < valid; c += 8) sum += __expf(Srow[c] - mx);
  __syncthreads();
  red[r * 8 + sub] = sum;
  __syncthreads();
  sum = 0.f;
#pragma unroll
  for (int j = 0; j < 8; ++j) sum += red[r * 8 + j];
  const float invs = 1.0f / sum;

  // write probs: fp32 to global (all 2048 cols incl. zeros), bf16 to LDS
  float*  arow   = attn_out + (bh * SEQ + (size_t)qt * 16 + r) * SEQ;
  bf16_t* prow_w = Pb + r * SEQ;
  for (int c = sub; c < SEQ; c += 8) {
    const float p = (c < valid) ? __expf(Srow[c] - mx) * invs : 0.f;
    arow[c]   = p;
    prow_w[c] = f2bf(p);
  }
  __syncthreads();

  // --- ctx = P[16, k] @ V[k, 64]; wave `wid` -> Dh cols wid*16..+15 ---
  const bf16_t* vrow = vT_ws + (bh * HEAD_DIM + (size_t)wid * 16 + nl) * SEQ;
  const bf16_t* prow = Pb + nl * SEQ;   // A: lane = query row m = nl
  v8f acc = {};
  const int nk = ((qt * 16 + 16) + 31) >> 5;   // K blocks of 32 (zero-padded)
  for (int kb = 0; kb < nk; ++kb) {
    v16bf pa = frag_bf16(prow, kb * 32, lane); // pure ds_load_b32 x8
    v16bf vb = frag_bf16(vrow, kb * 32, lane);
    acc = wmma_bf16(pa, vb, acc);
  }

  const size_t orow = (size_t)bi * SEQ + (size_t)qt * 16;
  const int col = h * HEAD_DIM + wid * 16 + nl;
#pragma unroll
  for (int v = 0; v < 8; ++v)
    ctx_ws[(orow + v + 8 * hl) * D_MODEL + col] = f2bf(acc[v]);
}

// ---------------------------------------------------------------------------
// Stage 3: x = ctx [8192,512] @ w_o^T + b_o  -> fp32 d_out[0 : 8192*512]
// grid (512, 8), block 128.  Both operands bf16.
// ---------------------------------------------------------------------------
__global__ __launch_bounds__(128) void out_proj_kernel(
    const bf16_t* __restrict__ A, const bf16_t* __restrict__ W,
    const float* __restrict__ bias, float* __restrict__ out) {
  const int lane = threadIdx.x & 31;
  const int wid  = threadIdx.x >> 5;
  const int m0 = blockIdx.x * 16;
  const int n0 = blockIdx.y * 64 + wid * 16;
  const int hl = (lane >> 4) & 1;
  const int nl = lane & 15;

  const bf16_t* arow = A + (size_t)(m0 + nl) * D_MODEL;
  const bf16_t* wrow = W + (size_t)(n0 + nl) * D_MODEL;

  v8f acc = {};
#pragma unroll 4
  for (int k0 = 0; k0 < D_MODEL; k0 += 32) {
    v16bf a = frag_bf16(arow, k0, lane);
    v16bf b = frag_bf16(wrow, k0, lane);
    acc = wmma_bf16(a, b, acc);
  }
  const float bv = bias[n0 + nl];
#pragma unroll
  for (int v = 0; v < 8; ++v)
    out[(size_t)(m0 + v + 8 * hl) * D_MODEL + n0 + nl] = acc[v] + bv;
}

// ---------------------------------------------------------------------------
extern "C" void kernel_launch(void* const* d_in, const int* in_sizes, int n_in,
                              void* d_out, int out_size, void* d_ws, size_t ws_size,
                              hipStream_t stream) {
  const float* Q  = (const float*)d_in[0];
  const float* K  = (const float*)d_in[1];
  const float* V  = (const float*)d_in[2];
  // d_in[3] = causal mask: handled analytically, unused
  const float* wq = (const float*)d_in[4];
  const float* bq = (const float*)d_in[5];
  const float* wk = (const float*)d_in[6];
  const float* bk = (const float*)d_in[7];
  const float* wv = (const float*)d_in[8];
  const float* bv = (const float*)d_in[9];
  const float* wo = (const float*)d_in[10];
  const float* bo = (const float*)d_in[11];

  const size_t MB = 1048576;
  char* ws = (char*)d_ws;
  bf16_t* Qb     = (bf16_t*)(ws);                  // 8 MB  [8192,512]
  bf16_t* Kb     = (bf16_t*)(ws + 8  * MB);        // 8 MB
  bf16_t* Vb     = (bf16_t*)(ws + 16 * MB);        // 8 MB
  bf16_t* wqb    = (bf16_t*)(ws + 24 * MB);        // 0.5 MB [512,512]
  bf16_t* wkb    = (bf16_t*)(ws + 24 * MB + 512 * 1024);
  bf16_t* wvb    = (bf16_t*)(ws + 25 * MB);
  bf16_t* wob    = (bf16_t*)(ws + 25 * MB + 512 * 1024);
  bf16_t* q_ws   = (bf16_t*)(ws + 26 * MB);        // 8 MB  [B,H,S,Dh]
  bf16_t* k_ws   = (bf16_t*)(ws + 34 * MB);        // 8 MB  [B,H,S,Dh]
  bf16_t* vT_ws  = (bf16_t*)(ws + 42 * MB);        // 8 MB  [B,H,Dh,S]
  bf16_t* ctx_ws = (bf16_t*)(ws + 50 * MB);        // 8 MB  [8192,512]

  float* x_out    = (float*)d_out;                   // [4,2048,512]
  float* attn_out = x_out + (size_t)MROWS * D_MODEL; // [4,8,2048,2048]

  // Stage 0: fp32 -> bf16 conversions (vectorized)
  const int nX = MROWS * D_MODEL / 4;   // float4s per activation tensor
  const int nW = D_MODEL * D_MODEL / 4; // float4s per weight matrix
  cvt_f32_bf16_kernel<<<(nX + 255) / 256, 256, 0, stream>>>(Q,  Qb,  nX);
  cvt_f32_bf16_kernel<<<(nX + 255) / 256, 256, 0, stream>>>(K,  Kb,  nX);
  cvt_f32_bf16_kernel<<<(nX + 255) / 256, 256, 0, stream>>>(V,  Vb,  nX);
  cvt_f32_bf16_kernel<<<(nW + 255) / 256, 256, 0, stream>>>(wq, wqb, nW);
  cvt_f32_bf16_kernel<<<(nW + 255) / 256, 256, 0, stream>>>(wk, wkb, nW);
  cvt_f32_bf16_kernel<<<(nW + 255) / 256, 256, 0, stream>>>(wv, wvb, nW);
  cvt_f32_bf16_kernel<<<(nW + 255) / 256, 256, 0, stream>>>(wo, wob, nW);

  // Stage 1: Q/K/V projections (fused over blockIdx.z)
  qkv_proj_kernel<<<dim3(MROWS / 16, D_MODEL / 64, 3), 128, 0, stream>>>(
      Qb, Kb, Vb, wqb, wkb, wvb, bq, bk, bv, q_ws, k_ws, vT_ws);

  // Stage 2: scores + softmax + prob output + ctx
  const size_t smem = (size_t)16 * SEQ * (sizeof(float) + sizeof(bf16_t))
                    + 128 * sizeof(float);           // ~192.5 KB
  attn_kernel<<<dim3(BH * (SEQ / 16)), 128, smem, stream>>>(
      q_ws, k_ws, vT_ws, attn_out, ctx_ws);

  // Stage 3: output projection
  out_proj_kernel<<<dim3(MROWS / 16, D_MODEL / 64), 128, 0, stream>>>(
      ctx_ws, wob, bo, x_out);
}
